// MAB_35064113005012
// MI455X (gfx1250) — matrix operand order repaired and checked by reading, not compiled, not run on hardware
//
#include <hip/hip_runtime.h>
#include <hip/hip_bf16.h>
#include <stdint.h>

// ---------------------------------------------------------------------------
// Types for CDNA5 WMMA (wave32, 16x16x32 bf16 -> f32)
// ---------------------------------------------------------------------------
typedef __attribute__((ext_vector_type(16))) __bf16 v16bf;
typedef __attribute__((ext_vector_type(8)))  float  v8f;
typedef unsigned short u16;

union BF16Tile {
  uint4 q[2];   // two 16-byte halves
  v16bf v;      // operand for v_wmma_f32_16x16x32_bf16
};

#define WMMA_BF16(A, B, C) \
  __builtin_amdgcn_wmma_f32_16x16x32_bf16(false, (A), false, (B), (short)0, (C), false, false)

// Scheduling fence: keep prefetch loads *issued* before the compute group so
// the scheduler cannot sink them back to just-before-use (which produces
// s_wait_loadcnt 0 stalls in front of every WMMA).
#if defined(__has_builtin)
#if __has_builtin(__builtin_amdgcn_sched_barrier)
#define SCHED_FENCE() __builtin_amdgcn_sched_barrier(0)
#endif
#endif
#ifndef SCHED_FENCE
#define SCHED_FENCE()
#endif

__device__ __forceinline__ u16 f2bf(float f) {
  union { float f; uint32_t u; } x; x.f = f;
  uint32_t u = x.u;
  uint32_t r = (u + 0x7FFFu + ((u >> 16) & 1u)) >> 16;  // RNE
  return (u16)r;
}
__device__ __forceinline__ float bf2f(u16 h) {
  union { uint32_t u; float f; } x; x.u = ((uint32_t)h) << 16;
  return x.f;
}

__device__ __forceinline__ float rowmax16(float x) {
  x = fmaxf(x, __shfl_xor(x, 1, 32));
  x = fmaxf(x, __shfl_xor(x, 2, 32));
  x = fmaxf(x, __shfl_xor(x, 4, 32));
  x = fmaxf(x, __shfl_xor(x, 8, 32));
  return x;
}
__device__ __forceinline__ float rowsum16(float x) {
  x += __shfl_xor(x, 1, 32);
  x += __shfl_xor(x, 2, 32);
  x += __shfl_xor(x, 4, 32);
  x += __shfl_xor(x, 8, 32);
  return x;
}

#define BQ     8
#define NQ     1024
#define DMODEL 512
#define HEADS  8
#define DHEAD  64
#define GM     (BQ * NQ)   // 8192 rows for all big GEMMs

// ---------------------------------------------------------------------------
// fp32 -> bf16 elementwise
// ---------------------------------------------------------------------------
__global__ __launch_bounds__(256) void cvt_bf16(const float* __restrict__ in,
                                                u16* __restrict__ out, int n) {
  int i = blockIdx.x * 256 + threadIdx.x;
  if (i < n) out[i] = f2bf(in[i]);
}

// Transpose-convert the four 512x512 weights: Wt[n][k] = W[k][n] (bf16)
__global__ __launch_bounds__(256) void cvt_w_t(
    const float* __restrict__ w0, const float* __restrict__ w1,
    const float* __restrict__ w2, const float* __restrict__ w3,
    u16* __restrict__ o0, u16* __restrict__ o1,
    u16* __restrict__ o2, u16* __restrict__ o3) {
  const float* w; u16* o;
  switch (blockIdx.y) {
    case 0: w = w0; o = o0; break;
    case 1: w = w1; o = o1; break;
    case 2: w = w2; o = o2; break;
    default: w = w3; o = o3; break;
  }
  int idx = blockIdx.x * 256 + threadIdx.x;   // 0 .. 512*512-1
  int n = idx >> 9, k = idx & 511;
  o[idx] = f2bf(w[k * DMODEL + n]);
}

// ---------------------------------------------------------------------------
// WMMA GEMM: C[8192,512] = A[8192,512] * Bt^T + bias
//   A  : bf16 row-major [M][K]
//   Bt : bf16 "B-transposed" [N][K]  (lane loads are contiguous b128s)
// 8 waves/block; wave tile 16(M) x 64(N); block tile 32 x 256.
// K loop is register double-buffered with a sched fence so next block's 10
// b128 loads are issued before the current block's 4 WMMAs.
// mode 0: store bf16 head-split  out[((b*H+h)*NQ+q)*64 + d]
// mode 1: store bf16 transposed  out[((b*H+h)*64+d)*NQ + q]   (packed 16B)
// mode 2: store f32 row-major    out[m*512 + n]
// ---------------------------------------------------------------------------
__global__ __launch_bounds__(256) void gemm_bf16_wmma(
    const u16* __restrict__ A, const u16* __restrict__ Bt,
    const float* __restrict__ bias, void* __restrict__ outp, int mode) {
  const int tid  = threadIdx.x;
  const int lane = tid & 31;
  const int wv   = tid >> 5;
  const int c16  = lane & 15;
  const int hf   = lane >> 4;
  const int mBase = blockIdx.x * 32 + (wv >> 2) * 16;
  const int nBase = blockIdx.y * 256 + (wv & 3) * 64;

  v8f acc0 = {}, acc1 = {}, acc2 = {}, acc3 = {};
  const u16* ap = A + (size_t)(mBase + c16) * DMODEL + hf * 16;
  const u16* bp = Bt + (size_t)(nBase + c16) * DMODEL + hf * 16;

  BF16Tile a[2], b0[2], b1[2], b2[2], b3[2];
  auto load_blk = [&](int kb, int s) {
    a[s].q[0]  = *(const uint4*)(ap + kb);
    a[s].q[1]  = *(const uint4*)(ap + kb + 8);
    b0[s].q[0] = *(const uint4*)(bp + kb);
    b0[s].q[1] = *(const uint4*)(bp + kb + 8);
    b1[s].q[0] = *(const uint4*)(bp + kb + 16 * DMODEL);
    b1[s].q[1] = *(const uint4*)(bp + kb + 16 * DMODEL + 8);
    b2[s].q[0] = *(const uint4*)(bp + kb + 32 * DMODEL);
    b2[s].q[1] = *(const uint4*)(bp + kb + 32 * DMODEL + 8);
    b3[s].q[0] = *(const uint4*)(bp + kb + 48 * DMODEL);
    b3[s].q[1] = *(const uint4*)(bp + kb + 48 * DMODEL + 8);
  };

  load_blk(0, 0);
#pragma unroll
  for (int kb = 0; kb < DMODEL; kb += 32) {
    const int s = (kb >> 5) & 1;         // constant after full unroll
    if (kb + 32 < DMODEL) load_blk(kb + 32, s ^ 1);
    SCHED_FENCE();                        // loads(k+1) stay above wmmas(k)
    acc0 = WMMA_BF16(a[s].v, b0[s].v, acc0);
    acc1 = WMMA_BF16(a[s].v, b1[s].v, acc1);
    acc2 = WMMA_BF16(a[s].v, b2[s].v, acc2);
    acc3 = WMMA_BF16(a[s].v, b3[s].v, acc3);
    SCHED_FENCE();                        // keep wmmas(k) above loads(k+2)
  }

  auto store_tile = [&](v8f acc, int t) {
    const int n = nBase + t * 16 + c16;     // global column (fixed per lane)
    const float bval = bias[n];
    if (mode == 0) {
      const int hh = n >> 6, d = n & 63;
      u16* o = (u16*)outp;
#pragma unroll
      for (int v = 0; v < 8; ++v) {
        const int m = mBase + hf * 8 + v;
        const int bb = m >> 10, qq = m & 1023;
        o[((size_t)(bb * HEADS + hh) * NQ + qq) * DHEAD + d] = f2bf(acc[v] + bval);
      }
    } else if (mode == 1) {
      const int hh = n >> 6, d = n & 63;
      const int m0 = mBase + hf * 8;
      const int bb = m0 >> 10, qq = m0 & 1023;
      u16 pk[8];
#pragma unroll
      for (int v = 0; v < 8; ++v) pk[v] = f2bf(acc[v] + bval);
      uint4 w;
      w.x = (uint32_t)pk[0] | ((uint32_t)pk[1] << 16);
      w.y = (uint32_t)pk[2] | ((uint32_t)pk[3] << 16);
      w.z = (uint32_t)pk[4] | ((uint32_t)pk[5] << 16);
      w.w = (uint32_t)pk[6] | ((uint32_t)pk[7] << 16);
      *(uint4*)((u16*)outp + ((size_t)(bb * HEADS + hh) * DHEAD + d) * NQ + qq) = w;
    } else {
      float* o = (float*)outp;
#pragma unroll
      for (int v = 0; v < 8; ++v)
        o[(size_t)(mBase + hf * 8 + v) * DMODEL + n] = acc[v] + bval;
    }
  };
  store_tile(acc0, 0); store_tile(acc1, 1); store_tile(acc2, 2); store_tile(acc3, 3);
}

// ---------------------------------------------------------------------------
// Flash attention: one wave owns 16 query rows, streams 32 keys / step.
// Pipelined: K tiles for chunk i+1, V tiles + mask bytes for chunk i are all
// issued before chunk i's math; a sched fence stops the scheduler from
// sinking them, so the softmax VALU stretch hides the load latency.
//   Qh [B,H,NQ,64] bf16 (A-operand), Kh [B,H,NQ,64] bf16 (== B^T of K^T),
//   Vt [B,H,64,NQ] bf16 (== B^T of V).  Output merged heads: Obf[B*NQ][512].
// ---------------------------------------------------------------------------
__global__ __launch_bounds__(128) void attn_flash(
    const u16* __restrict__ Qh, const u16* __restrict__ Kh,
    const u16* __restrict__ Vt, const unsigned char* __restrict__ mask,
    u16* __restrict__ Obf) {
  __shared__ __align__(16) u16 lds[4 * 16 * 32];
  const int tid  = threadIdx.x;
  const int lane = tid & 31;
  const int wv   = tid >> 5;
  const int c16  = lane & 15;
  const int hf   = lane >> 4;
  const int bh = blockIdx.x >> 4;   // 0..63 (b*8+h)
  const int rb = blockIdx.x & 15;   // 64-row block within head
  const int b  = bh >> 3;
  const int h  = bh & 7;
  const int q0 = rb * 64 + wv * 16;
  const u16* Qp = Qh + (size_t)bh * NQ * DHEAD;
  const u16* Kp = Kh + (size_t)bh * NQ * DHEAD;
  const u16* Vp = Vt + (size_t)bh * DHEAD * NQ;
  const unsigned char* mp = mask + b * NQ;
  u16* lp = lds + wv * (16 * 32);

  // Q resident as two A tiles (features 0..31 and 32..63)
  BF16Tile aq0, aq1;
  {
    const u16* p = Qp + (size_t)(q0 + c16) * DHEAD;
    aq0.q[0] = *(const uint4*)(p + hf * 16);
    aq0.q[1] = *(const uint4*)(p + 8 + hf * 16);
    aq1.q[0] = *(const uint4*)(p + 32 + hf * 16);
    aq1.q[1] = *(const uint4*)(p + 40 + hf * 16);
  }

  v8f o0 = {}, o1 = {}, o2 = {}, o3 = {};
  float mrun[8], lrun[8];
#pragma unroll
  for (int v = 0; v < 8; ++v) { mrun[v] = -1e30f; lrun[v] = 0.f; }
  const float scale = 1.25f;  // 1/(sqrt(64)*0.1)

  // ---- K tile loader (4 tiles = 32 keys x 64 features)
  auto load_k = [&](int kb, BF16Tile& K0, BF16Tile& K1, BF16Tile& K2, BF16Tile& K3) {
    const u16* bp0 = Kp + (size_t)(kb + c16) * DHEAD + hf * 16;
    K0.q[0] = *(const uint4*)(bp0);      K0.q[1] = *(const uint4*)(bp0 + 8);
    K1.q[0] = *(const uint4*)(bp0 + 32); K1.q[1] = *(const uint4*)(bp0 + 40);
    const u16* bp1 = Kp + (size_t)(kb + 16 + c16) * DHEAD + hf * 16;
    K2.q[0] = *(const uint4*)(bp1);      K2.q[1] = *(const uint4*)(bp1 + 8);
    K3.q[0] = *(const uint4*)(bp1 + 32); K3.q[1] = *(const uint4*)(bp1 + 40);
  };

  // ---- one 32-key flash step using already-resident K tiles
  auto step = [&](int kb, const BF16Tile& K0, const BF16Tile& K1,
                  const BF16Tile& K2, const BF16Tile& K3) {
    // V tiles + mask bytes issued first: consumed only after the softmax.
    BF16Tile v0t, v1t, v2t, v3t;
    {
      const u16* vp = Vp + (size_t)c16 * NQ + kb + hf * 16;
      v0t.q[0] = *(const uint4*)(vp);
      v0t.q[1] = *(const uint4*)(vp + 8);
      v1t.q[0] = *(const uint4*)(vp + 16 * NQ);
      v1t.q[1] = *(const uint4*)(vp + 16 * NQ + 8);
      v2t.q[0] = *(const uint4*)(vp + 32 * NQ);
      v2t.q[1] = *(const uint4*)(vp + 32 * NQ + 8);
      v3t.q[0] = *(const uint4*)(vp + 48 * NQ);
      v3t.q[1] = *(const uint4*)(vp + 48 * NQ + 8);
    }
    const float msk0 = mp[kb + c16]      ? 0.f : -1e30f;
    const float msk1 = mp[kb + 16 + c16] ? 0.f : -1e30f;
    SCHED_FENCE();   // all loads above stay issued before the math below

    // S = Q K^T (16 x 32): two 16x16 subtiles, K-dim 64 = 2 WMMAs each
    v8f s0 = {}, s1 = {};
    s0 = WMMA_BF16(aq0.v, K0.v, s0);
    s0 = WMMA_BF16(aq1.v, K1.v, s0);
    s1 = WMMA_BF16(aq0.v, K2.v, s1);
    s1 = WMMA_BF16(aq1.v, K3.v, s1);

    // online softmax (each C-layout row lives in one 16-lane half)
    float al[8];
#pragma unroll
    for (int v = 0; v < 8; ++v) {
      float x0 = s0[v] * scale + msk0;
      float x1 = s1[v] * scale + msk1;
      float t  = rowmax16(fmaxf(x0, x1));
      float mn = fmaxf(mrun[v], t);
      al[v] = __expf(mrun[v] - mn);
      mrun[v] = mn;
      float p0 = __expf(x0 - mn);
      float p1 = __expf(x1 - mn);
      s0[v] = p0; s1[v] = p1;
      lrun[v] = lrun[v] * al[v] + rowsum16(p0 + p1);
    }
#pragma unroll
    for (int v = 0; v < 8; ++v) {
      o0[v] *= al[v]; o1[v] *= al[v]; o2[v] *= al[v]; o3[v] *= al[v];
    }

    // transpose P from C-layout to A-layout via per-wave LDS slice
#pragma unroll
    for (int v = 0; v < 8; ++v) {
      const int r = hf * 8 + v;
      lp[r * 32 + c16]      = f2bf(s0[v]);
      lp[r * 32 + 16 + c16] = f2bf(s1[v]);
    }
    BF16Tile pa;
    {
      const u16* lr = lp + c16 * 32;
      pa.q[0] = *(const uint4*)(lr + hf * 16);
      pa.q[1] = *(const uint4*)(lr + 8 + hf * 16);
    }

    // O += P * V   (K=32, 4 N-tiles over d_head=64)
    o0 = WMMA_BF16(pa.v, v0t.v, o0);
    o1 = WMMA_BF16(pa.v, v1t.v, o1);
    o2 = WMMA_BF16(pa.v, v2t.v, o2);
    o3 = WMMA_BF16(pa.v, v3t.v, o3);
  };

  // ---- ping-pong over 64-key super-iterations
  BF16Tile ka0, ka1, ka2, ka3;   // even chunk K tiles
  BF16Tile kb0, kb1, kb2, kb3;   // odd chunk K tiles
  load_k(0, ka0, ka1, ka2, ka3);
  for (int kb = 0; kb < NQ; kb += 64) {
    load_k(kb + 32, kb0, kb1, kb2, kb3);     // prefetch odd chunk
    step(kb, ka0, ka1, ka2, ka3);
    if (kb + 64 < NQ) load_k(kb + 64, ka0, ka1, ka2, ka3);  // prefetch next even
    step(kb + 32, kb0, kb1, kb2, kb3);
  }

  // ---- finalize and store merged-head bf16 output
#pragma unroll
  for (int v = 0; v < 8; ++v) {
    const float inv = 1.f / fmaxf(lrun[v], 1e-20f);
    o0[v] *= inv; o1[v] *= inv; o2[v] *= inv; o3[v] *= inv;
  }
  const int rowbase = b * NQ + q0 + hf * 8;
#pragma unroll
  for (int v = 0; v < 8; ++v) {
    u16* orow = Obf + (size_t)(rowbase + v) * DMODEL + h * DHEAD + c16;
    orow[0]  = f2bf(o0[v]);
    orow[16] = f2bf(o1[v]);
    orow[32] = f2bf(o2[v]);
    orow[48] = f2bf(o3[v]);
  }
}

// ---------------------------------------------------------------------------
// Residual + ReLU + LayerNorm: out = LN(O + relu(G)) * gamma + beta
//   O: bf16 [8192][512], G = O@Wo + bo : f32 [8192][512]
// ---------------------------------------------------------------------------
__global__ __launch_bounds__(256) void ln_kernel(
    const u16* __restrict__ Obf, const float* __restrict__ G,
    const float* __restrict__ gamma, const float* __restrict__ beta,
    float* __restrict__ out) {
  const int row = blockIdx.x;
  const int tid = threadIdx.x;
  const u16* op = Obf + (size_t)row * DMODEL;
  const float* gp = G + (size_t)row * DMODEL;
  const int j0 = tid, j1 = tid + 256;
  float y0 = bf2f(op[j0]) + fmaxf(gp[j0], 0.f);
  float y1 = bf2f(op[j1]) + fmaxf(gp[j1], 0.f);
  float s  = y0 + y1;
  float sq = y0 * y0 + y1 * y1;
#pragma unroll
  for (int m = 1; m < 32; m <<= 1) {
    s  += __shfl_xor(s, m, 32);
    sq += __shfl_xor(sq, m, 32);
  }
  __shared__ float ss[8], sqq[8];
  const int wvi = tid >> 5;
  if ((tid & 31) == 0) { ss[wvi] = s; sqq[wvi] = sq; }
  __syncthreads();
  s = 0.f; sq = 0.f;
#pragma unroll
  for (int i = 0; i < 8; ++i) { s += ss[i]; sq += sqq[i]; }
  const float mu  = s * (1.f / DMODEL);
  const float var = sq * (1.f / DMODEL) - mu * mu;
  const float r   = rsqrtf(var + 1e-5f);
  out[(size_t)row * DMODEL + j0] = (y0 - mu) * r * gamma[j0] + beta[j0];
  out[(size_t)row * DMODEL + j1] = (y1 - mu) * r * gamma[j1] + beta[j1];
}

// ---------------------------------------------------------------------------
// Launch
// ---------------------------------------------------------------------------
extern "C" void kernel_launch(void* const* d_in, const int* in_sizes, int n_in,
                              void* d_out, int out_size, void* d_ws, size_t ws_size,
                              hipStream_t stream) {
  (void)in_sizes; (void)n_in; (void)out_size; (void)ws_size;
  const float* Q  = (const float*)d_in[0];
  const float* K  = (const float*)d_in[1];
  const unsigned char* mask = (const unsigned char*)d_in[2];
  const float* Wq = (const float*)d_in[3];
  const float* bq = (const float*)d_in[4];
  const float* Wk = (const float*)d_in[5];
  const float* bk = (const float*)d_in[6];
  const float* Wv = (const float*)d_in[7];
  const float* bv = (const float*)d_in[8];
  const float* Wo = (const float*)d_in[9];
  const float* bo = (const float*)d_in[10];
  const float* gamma = (const float*)d_in[11];
  const float* beta  = (const float*)d_in[12];
  float* out = (float*)d_out;

  char* ws = (char*)d_ws;
  size_t off = 0;
  auto alloc = [&](size_t bytes) {
    char* p = ws + off;
    off += (bytes + 255) & ~(size_t)255;
    return p;
  };
  u16* Qbf  = (u16*)alloc((size_t)GM * DMODEL * 2);
  u16* Kbf  = (u16*)alloc((size_t)GM * DMODEL * 2);
  u16* Wqt  = (u16*)alloc((size_t)DMODEL * DMODEL * 2);
  u16* Wkt  = (u16*)alloc((size_t)DMODEL * DMODEL * 2);
  u16* Wvt  = (u16*)alloc((size_t)DMODEL * DMODEL * 2);
  u16* Wot  = (u16*)alloc((size_t)DMODEL * DMODEL * 2);
  u16* Qhp  = (u16*)alloc((size_t)GM * DMODEL * 2);
  u16* Khp  = (u16*)alloc((size_t)GM * DMODEL * 2);
  u16* Vtp  = (u16*)alloc((size_t)GM * DMODEL * 2);
  u16* Obfp = (u16*)alloc((size_t)GM * DMODEL * 2);
  float* G  = (float*)alloc((size_t)GM * DMODEL * 4);

  const int NE = GM * DMODEL;  // 4,194,304
  cvt_bf16<<<NE / 256, 256, 0, stream>>>(Q, Qbf, NE);
  cvt_bf16<<<NE / 256, 256, 0, stream>>>(K, Kbf, NE);
  cvt_w_t<<<dim3((DMODEL * DMODEL) / 256, 4), 256, 0, stream>>>(
      Wq, Wk, Wv, Wo, Wqt, Wkt, Wvt, Wot);

  dim3 gg(GM / 32, DMODEL / 256);
  gemm_bf16_wmma<<<gg, 256, 0, stream>>>(Qbf, Wqt, bq, Qhp, 0);
  gemm_bf16_wmma<<<gg, 256, 0, stream>>>(Kbf, Wkt, bk, Khp, 0);
  gemm_bf16_wmma<<<gg, 256, 0, stream>>>(Kbf, Wvt, bv, Vtp, 1);

  attn_flash<<<BQ * HEADS * (NQ / 64), 128, 0, stream>>>(Qhp, Khp, Vtp, mask, Obfp);

  gemm_bf16_wmma<<<gg, 256, 0, stream>>>(Obfp, Wot, bo, G, 2);
  ln_kernel<<<GM, 256, 0, stream>>>(Obfp, G, gamma, beta, out);
}
